// AdHocWolfpackGNN_82343112999098
// MI455X (gfx1250) — matrix operand order repaired and checked by reading, not compiled, not run on hardware
//
#include <hip/hip_runtime.h>
#include <cstdint>
#include <cstddef>

typedef __attribute__((ext_vector_type(16))) _Float16 v16h;
typedef __attribute__((ext_vector_type(8)))  float    v8f;

#define Bg 16
#define Nn 2048
#define Ee 65536
#define Hh 32
#define ETc (Bg*Ee)   /* 1048576 */
#define NTc (Bg*Nn)   /* 32768   */

// ---- d_out offsets (float elements, reference return order) ----
#define OFF_EH   ((size_t)ETc*Hh)
#define OFF_EC   ((size_t)2*ETc*Hh)
#define OFF_NOUT ((size_t)3*ETc*Hh)
#define OFF_NH   (OFF_NOUT + (size_t)NTc*Hh)
#define OFF_NC   (OFF_NOUT + (size_t)2*NTc*Hh)
#define OFF_UOUT (OFF_NOUT + (size_t)3*NTc*Hh)
#define OFF_UH   (OFF_UOUT + (size_t)Bg*Hh)
#define OFF_UC   (OFF_UOUT + (size_t)2*Bg*Hh)

// ---- workspace byte offsets ----
#define WS_FRAGE 0u                                   /* 5*8*32*16 halfs = 40960 B */
#define WS_FRAGN 40960u                               /* 4*8*32*16 halfs = 32768 B */
#define WS_FRAGU 73728u                               /* 32768 B */
#define WS_BIASE 106496u                              /* 128 f32 */
#define WS_BIASN 107008u
#define WS_BIASU 107520u
#define WS_HAGG  108544u                              /* NT*H f32 = 4 MB */
#define WS_ECOMB (WS_HAGG + (size_t)NTc*Hh*4)
#define WS_NCOMB (WS_ECOMB + (size_t)Bg*Hh*4)

__device__ __forceinline__ float sigf(float x) { return 1.0f / (1.0f + __expf(-x)); }

__device__ __forceinline__ void atomAddG(float* p, float v) {
    __hip_atomic_fetch_add(p, v, __ATOMIC_RELAXED, __HIP_MEMORY_SCOPE_AGENT);
}
__device__ __forceinline__ void atomAddS(float* p, float v) {
    __hip_atomic_fetch_add(p, v, __ATOMIC_RELAXED, __HIP_MEMORY_SCOPE_WORKGROUP);
}

// Build an f16 A-fragment (16x32 MxK tile) for one K-block from a 32-wide f32 row.
// Lane layout: lanes<16 -> K {0..15}; lanes>=16 -> K {16..31}.
__device__ __forceinline__ v16h load_a(const float* __restrict__ p, int lo) {
    const float4* q = (const float4*)p;
    const int b0 = lo >> 1;                 // 0 or 4
    float4 x0 = q[b0],     x1 = q[b0 + 1];  // K = 2*lo..2*lo+7
    float4 x2 = q[b0 + 2], x3 = q[b0 + 3];  // K = 2*lo+8..2*lo+15
    v16h r;
    r[0]=(_Float16)x0.x; r[1]=(_Float16)x0.y; r[2]=(_Float16)x0.z; r[3]=(_Float16)x0.w;
    r[4]=(_Float16)x1.x; r[5]=(_Float16)x1.y; r[6]=(_Float16)x1.z; r[7]=(_Float16)x1.w;
    r[8]=(_Float16)x2.x; r[9]=(_Float16)x2.y; r[10]=(_Float16)x2.z; r[11]=(_Float16)x2.w;
    r[12]=(_Float16)x3.x; r[13]=(_Float16)x3.y; r[14]=(_Float16)x3.z; r[15]=(_Float16)x3.w;
    return r;
}

// ---------------------------------------------------------------------------
// Zero the accumulation buffers in the workspace (h_agg, e_comb, n_comb).
// ---------------------------------------------------------------------------
__global__ void zero_kernel(float* __restrict__ hagg, float* __restrict__ ecomb,
                            float* __restrict__ ncomb) {
    const int i = blockIdx.x * blockDim.x + threadIdx.x;
    if (i < NTc * Hh) hagg[i] = 0.0f;
    if (i < Bg * Hh) { ecomb[i] = 0.0f; ncomb[i] = 0.0f; }
}

// ---------------------------------------------------------------------------
// Pack [W_ih | W_hh] (f32) into f16 WMMA B-fragments + fused bias.
// Fragment f = s*8+t: lane<16 -> column n=16t+lane, halves j -> K=32s+j;
//                     lane>=16 -> same column, halves j -> K=32s+16+j.
// ---------------------------------------------------------------------------
__global__ __launch_bounds__(256) void pack_kernel(
    const float* __restrict__ We_ih, const float* __restrict__ We_hh,
    const float* __restrict__ be_ih, const float* __restrict__ be_hh,
    const float* __restrict__ Wn_ih, const float* __restrict__ Wn_hh,
    const float* __restrict__ bn_ih, const float* __restrict__ bn_hh,
    const float* __restrict__ Wu_ih, const float* __restrict__ Wu_hh,
    const float* __restrict__ bu_ih, const float* __restrict__ bu_hh,
    _Float16* __restrict__ fragE, _Float16* __restrict__ fragN,
    _Float16* __restrict__ fragU, float* __restrict__ biasE,
    float* __restrict__ biasN, float* __restrict__ biasU) {
    const int tid = threadIdx.x;
    // edge: Kcat=160 (ih 128 + hh 32), 5 K-steps
    for (int idx = tid; idx < 5 * 8 * 32 * 16; idx += 256) {
        const int f = idx >> 9, lane = (idx >> 4) & 31, j = idx & 15;
        const int s = f >> 3, t = f & 7;
        const int n = t * 16 + (lane & 15);
        const int k = s * 32 + ((lane & 16) ? 16 : 0) + j;
        const float v = (k < 128) ? We_ih[n * 128 + k] : We_hh[n * 32 + (k - 128)];
        fragE[idx] = (_Float16)v;
    }
    // node: Kcat=128 (ih 96 + hh 32), 4 K-steps
    for (int idx = tid; idx < 4 * 8 * 32 * 16; idx += 256) {
        const int f = idx >> 9, lane = (idx >> 4) & 31, j = idx & 15;
        const int s = f >> 3, t = f & 7;
        const int n = t * 16 + (lane & 15);
        const int k = s * 32 + ((lane & 16) ? 16 : 0) + j;
        const float v = (k < 96) ? Wn_ih[n * 96 + k] : Wn_hh[n * 32 + (k - 96)];
        fragN[idx] = (_Float16)v;
    }
    // graph: Kcat=128 (ih 96 + hh 32), 4 K-steps
    for (int idx = tid; idx < 4 * 8 * 32 * 16; idx += 256) {
        const int f = idx >> 9, lane = (idx >> 4) & 31, j = idx & 15;
        const int s = f >> 3, t = f & 7;
        const int n = t * 16 + (lane & 15);
        const int k = s * 32 + ((lane & 16) ? 16 : 0) + j;
        const float v = (k < 96) ? Wu_ih[n * 96 + k] : Wu_hh[n * 32 + (k - 96)];
        fragU[idx] = (_Float16)v;
    }
    for (int n = tid; n < 128; n += 256) {
        biasE[n] = be_ih[n] + be_hh[n];
        biasN[n] = bn_ih[n] + bn_hh[n];
        biasU[n] = bu_ih[n] + bu_hh[n];
    }
}

// ---------------------------------------------------------------------------
// Edge LSTM: 2048 blocks x 8 waves; each wave: 16-edge tile, 40 WMMAs.
// Blocks are graph-aligned (128 blocks/graph, 512 edges/block, 4 iters).
// ---------------------------------------------------------------------------
__global__ __launch_bounds__(256) void edge_lstm_kernel(
    const float* __restrict__ edge_feat, const float* __restrict__ node_feat,
    const float* __restrict__ g_repr, const float* __restrict__ edge_h1,
    const float* __restrict__ edge_h2, const int* __restrict__ edge_src,
    const int* __restrict__ edge_dst, const _Float16* __restrict__ fragE,
    const float* __restrict__ biasE, float* __restrict__ hagg,
    float* __restrict__ ecomb, float* __restrict__ out) {
    __shared__ alignas(32) _Float16 smemB[5 * 8 * 32 * 16];
    __shared__ float sbias[128];
    __shared__ float scomb[32];
    const int tid = threadIdx.x;
    const int lane = tid & 31;
    const int w = tid >> 5;
    {   // stage packed weights (40 KB) into LDS once per block
        const uint4* s = (const uint4*)fragE;
        uint4* d = (uint4*)smemB;
#pragma unroll
        for (int i = 0; i < 10; ++i) d[tid + 256 * i] = s[tid + 256 * i];
    }
    if (tid < 128) sbias[tid] = biasE[tid];
    if (tid < 32) scomb[tid] = 0.0f;
    __syncthreads();

    const int blk = blockIdx.x;
    const int g = blk >> 7;                 // graph id (128 blocks per graph)
    const int blk_base = blk * 512;         // first edge of this block
    const int jlo = lane & 15;
    const int lo = (lane & 16) ? 8 : 0;
    const int hi8 = (lane & 16) ? 8 : 0;
    const v16h* Bf = (const v16h*)smemB;
    const float* grow = g_repr + g * Hh;

    const float bi0 = sbias[jlo],      bi1 = sbias[16 + jlo];
    const float bf0 = sbias[32 + jlo], bf1 = sbias[48 + jlo];
    const float bg0 = sbias[64 + jlo], bg1 = sbias[80 + jlo];
    const float bo0 = sbias[96 + jlo], bo1 = sbias[112 + jlo];

    float sum_lo = 0.0f, sum_hi = 0.0f;

    for (int it = 0; it < 4; ++it) {
        const int base = blk_base + it * 128 + w * 16;
        // ---- build A fragments: concat(edge_feat, nf[src], nf[dst], u, h1) ----
        const int ea = base + jlo;
        const int sA = edge_src[ea] + g * Nn;
        const int dA = edge_dst[ea] + g * Nn;
        v16h As[5];
        As[0] = load_a(edge_feat + (size_t)ea * Hh, lo);
        As[1] = load_a(node_feat + (size_t)sA * Hh, lo);
        As[2] = load_a(node_feat + (size_t)dA * Hh, lo);
        As[3] = load_a(grow, lo);
        As[4] = load_a(edge_h1 + (size_t)ea * Hh, lo);

        v8f acc[8] = {};
#pragma unroll
        for (int s = 0; s < 5; ++s) {
#pragma unroll
            for (int t = 0; t < 8; ++t) {
                acc[t] = __builtin_amdgcn_wmma_f32_16x16x32_f16(
                    false, As[s], false, Bf[(s * 8 + t) * 32 + lane],
                    (short)0, acc[t], false, false);
            }
        }
        // ---- LSTM epilogue: tiles {0,1}=i {2,3}=f {4,5}=g {6,7}=o ----
#pragma unroll
        for (int r = 0; r < 8; ++r) {
            const int ee = base + r + hi8;
            const size_t rowo = (size_t)ee * Hh;
            const int dn = edge_dst[ee] + g * Nn;
            {
                const float ii = acc[0][r] + bi0, ff = acc[2][r] + bf0;
                const float gg = acc[4][r] + bg0, oo = acc[6][r] + bo0;
                const float cp = edge_h2[rowo + jlo];
                const float c2 = sigf(ff) * cp + sigf(ii) * tanhf(gg);
                const float h2 = sigf(oo) * tanhf(c2);
                const float eo = fmaxf(h2, 0.0f);
                out[rowo + jlo] = eo;
                out[OFF_EH + rowo + jlo] = h2;
                out[OFF_EC + rowo + jlo] = c2;
                atomAddG(&hagg[(size_t)dn * Hh + jlo], eo);
                sum_lo += eo;
            }
            {
                const int j = 16 + jlo;
                const float ii = acc[1][r] + bi1, ff = acc[3][r] + bf1;
                const float gg = acc[5][r] + bg1, oo = acc[7][r] + bo1;
                const float cp = edge_h2[rowo + j];
                const float c2 = sigf(ff) * cp + sigf(ii) * tanhf(gg);
                const float h2 = sigf(oo) * tanhf(c2);
                const float eo = fmaxf(h2, 0.0f);
                out[rowo + j] = eo;
                out[OFF_EH + rowo + j] = h2;
                out[OFF_EC + rowo + j] = c2;
                atomAddG(&hagg[(size_t)dn * Hh + j], eo);
                sum_hi += eo;
            }
        }
    }
    // per-graph readout partials: lane j-slots -> LDS -> one global atomic/slot
    atomAddS(&scomb[jlo], sum_lo);
    atomAddS(&scomb[16 + jlo], sum_hi);
    __syncthreads();
    if (tid < 32) atomAddG(&ecomb[g * Hh + tid], scomb[tid]);
}

// ---------------------------------------------------------------------------
// Node LSTM: 128 blocks (8/graph), 256 nodes/block, 2 iters; 32 WMMAs/tile.
// ---------------------------------------------------------------------------
__global__ __launch_bounds__(256) void node_lstm_kernel(
    const float* __restrict__ node_feat, const float* __restrict__ g_repr,
    const float* __restrict__ node_h1, const float* __restrict__ node_h2,
    const float* __restrict__ hagg, const _Float16* __restrict__ fragN,
    const float* __restrict__ biasN, float* __restrict__ ncomb,
    float* __restrict__ out) {
    __shared__ alignas(32) _Float16 smemB[4 * 8 * 32 * 16];
    __shared__ float sbias[128];
    __shared__ float scomb[32];
    const int tid = threadIdx.x;
    const int lane = tid & 31;
    const int w = tid >> 5;
    {
        const uint4* s = (const uint4*)fragN;
        uint4* d = (uint4*)smemB;
#pragma unroll
        for (int i = 0; i < 8; ++i) d[tid + 256 * i] = s[tid + 256 * i];
    }
    if (tid < 128) sbias[tid] = biasN[tid];
    if (tid < 32) scomb[tid] = 0.0f;
    __syncthreads();

    const int blk = blockIdx.x;
    const int g = blk >> 3;                 // 8 blocks per graph
    const int blk_base = blk * 256;
    const int jlo = lane & 15;
    const int lo = (lane & 16) ? 8 : 0;
    const int hi8 = (lane & 16) ? 8 : 0;
    const v16h* Bf = (const v16h*)smemB;
    const float* grow = g_repr + g * Hh;

    const float bi0 = sbias[jlo],      bi1 = sbias[16 + jlo];
    const float bf0 = sbias[32 + jlo], bf1 = sbias[48 + jlo];
    const float bg0 = sbias[64 + jlo], bg1 = sbias[80 + jlo];
    const float bo0 = sbias[96 + jlo], bo1 = sbias[112 + jlo];

    float sum_lo = 0.0f, sum_hi = 0.0f;

    for (int it = 0; it < 2; ++it) {
        const int base = blk_base + it * 128 + w * 16;
        const int na = base + jlo;
        v16h As[4];
        As[0] = load_a(node_feat + (size_t)na * Hh, lo);
        As[1] = load_a(hagg + (size_t)na * Hh, lo);
        As[2] = load_a(grow, lo);
        As[3] = load_a(node_h1 + (size_t)na * Hh, lo);

        v8f acc[8] = {};
#pragma unroll
        for (int s = 0; s < 4; ++s) {
#pragma unroll
            for (int t = 0; t < 8; ++t) {
                acc[t] = __builtin_amdgcn_wmma_f32_16x16x32_f16(
                    false, As[s], false, Bf[(s * 8 + t) * 32 + lane],
                    (short)0, acc[t], false, false);
            }
        }
#pragma unroll
        for (int r = 0; r < 8; ++r) {
            const int nn = base + r + hi8;
            const size_t rowo = (size_t)nn * Hh;
            {
                const float ii = acc[0][r] + bi0, ff = acc[2][r] + bf0;
                const float gg = acc[4][r] + bg0, oo = acc[6][r] + bo0;
                const float cp = node_h2[rowo + jlo];
                const float c2 = sigf(ff) * cp + sigf(ii) * tanhf(gg);
                const float h2 = sigf(oo) * tanhf(c2);
                const float no = fmaxf(h2, 0.0f);
                out[OFF_NOUT + rowo + jlo] = no;
                out[OFF_NH + rowo + jlo] = h2;
                out[OFF_NC + rowo + jlo] = c2;
                sum_lo += no;
            }
            {
                const int j = 16 + jlo;
                const float ii = acc[1][r] + bi1, ff = acc[3][r] + bf1;
                const float gg = acc[5][r] + bg1, oo = acc[7][r] + bo1;
                const float cp = node_h2[rowo + j];
                const float c2 = sigf(ff) * cp + sigf(ii) * tanhf(gg);
                const float h2 = sigf(oo) * tanhf(c2);
                const float no = fmaxf(h2, 0.0f);
                out[OFF_NOUT + rowo + j] = no;
                out[OFF_NH + rowo + j] = h2;
                out[OFF_NC + rowo + j] = c2;
                sum_hi += no;
            }
        }
    }
    atomAddS(&scomb[jlo], sum_lo);
    atomAddS(&scomb[16 + jlo], sum_hi);
    __syncthreads();
    if (tid < 32) atomAddG(&ncomb[g * Hh + tid], scomb[tid]);
}

// ---------------------------------------------------------------------------
// Graph LSTM: one wave, M=16 graphs exactly, 32 WMMAs.
// ---------------------------------------------------------------------------
__global__ __launch_bounds__(32) void graph_lstm_kernel(
    const float* __restrict__ g_repr, const float* __restrict__ graph_h1,
    const float* __restrict__ graph_h2, const float* __restrict__ ecomb,
    const float* __restrict__ ncomb, const _Float16* __restrict__ fragU,
    const float* __restrict__ biasU, float* __restrict__ out) {
    const int lane = threadIdx.x;
    const int jlo = lane & 15;
    const int lo = (lane & 16) ? 8 : 0;
    const int hi8 = (lane & 16) ? 8 : 0;
    const v16h* Bf = (const v16h*)fragU;

    v16h As[4];
    As[0] = load_a(ncomb + (size_t)jlo * Hh, lo);
    As[1] = load_a(ecomb + (size_t)jlo * Hh, lo);
    As[2] = load_a(g_repr + (size_t)jlo * Hh, lo);
    As[3] = load_a(graph_h1 + (size_t)jlo * Hh, lo);

    v8f acc[8] = {};
#pragma unroll
    for (int s = 0; s < 4; ++s) {
#pragma unroll
        for (int t = 0; t < 8; ++t) {
            acc[t] = __builtin_amdgcn_wmma_f32_16x16x32_f16(
                false, As[s], false, Bf[(s * 8 + t) * 32 + lane],
                (short)0, acc[t], false, false);
        }
    }
    const float bi0 = biasU[jlo],      bi1 = biasU[16 + jlo];
    const float bf0 = biasU[32 + jlo], bf1 = biasU[48 + jlo];
    const float bg0 = biasU[64 + jlo], bg1 = biasU[80 + jlo];
    const float bo0 = biasU[96 + jlo], bo1 = biasU[112 + jlo];
#pragma unroll
    for (int r = 0; r < 8; ++r) {
        const int bb = r + hi8;                // graph index
        const size_t rowo = (size_t)bb * Hh;
        {
            const float ii = acc[0][r] + bi0, ff = acc[2][r] + bf0;
            const float gg = acc[4][r] + bg0, oo = acc[6][r] + bo0;
            const float cp = graph_h2[rowo + jlo];
            const float c2 = sigf(ff) * cp + sigf(ii) * tanhf(gg);
            const float h2 = sigf(oo) * tanhf(c2);
            out[OFF_UOUT + rowo + jlo] = fmaxf(h2, 0.0f);
            out[OFF_UH + rowo + jlo] = h2;
            out[OFF_UC + rowo + jlo] = c2;
        }
        {
            const int j = 16 + jlo;
            const float ii = acc[1][r] + bi1, ff = acc[3][r] + bf1;
            const float gg = acc[5][r] + bg1, oo = acc[7][r] + bo1;
            const float cp = graph_h2[rowo + j];
            const float c2 = sigf(ff) * cp + sigf(ii) * tanhf(gg);
            const float h2 = sigf(oo) * tanhf(c2);
            out[OFF_UOUT + rowo + j] = fmaxf(h2, 0.0f);
            out[OFF_UH + rowo + j] = h2;
            out[OFF_UC + rowo + j] = c2;
        }
    }
}

extern "C" void kernel_launch(void* const* d_in, const int* in_sizes, int n_in,
                              void* d_out, int out_size, void* d_ws, size_t ws_size,
                              hipStream_t stream) {
    (void)in_sizes; (void)n_in; (void)out_size; (void)ws_size;
    const float* edge_feat = (const float*)d_in[0];
    const float* node_feat = (const float*)d_in[1];
    const float* g_repr    = (const float*)d_in[2];
    const float* edge_h1   = (const float*)d_in[3];
    const float* edge_h2   = (const float*)d_in[4];
    const float* node_h1   = (const float*)d_in[5];
    const float* node_h2   = (const float*)d_in[6];
    const float* graph_h1  = (const float*)d_in[7];
    const float* graph_h2  = (const float*)d_in[8];
    const int*   edge_src  = (const int*)d_in[9];
    const int*   edge_dst  = (const int*)d_in[10];
    const float* We_ih = (const float*)d_in[11];
    const float* We_hh = (const float*)d_in[12];
    const float* be_ih = (const float*)d_in[13];
    const float* be_hh = (const float*)d_in[14];
    const float* Wn_ih = (const float*)d_in[15];
    const float* Wn_hh = (const float*)d_in[16];
    const float* bn_ih = (const float*)d_in[17];
    const float* bn_hh = (const float*)d_in[18];
    const float* Wu_ih = (const float*)d_in[19];
    const float* Wu_hh = (const float*)d_in[20];
    const float* bu_ih = (const float*)d_in[21];
    const float* bu_hh = (const float*)d_in[22];

    char* ws = (char*)d_ws;
    _Float16* fragE = (_Float16*)(ws + WS_FRAGE);
    _Float16* fragN = (_Float16*)(ws + WS_FRAGN);
    _Float16* fragU = (_Float16*)(ws + WS_FRAGU);
    float* biasE = (float*)(ws + WS_BIASE);
    float* biasN = (float*)(ws + WS_BIASN);
    float* biasU = (float*)(ws + WS_BIASU);
    float* hagg  = (float*)(ws + WS_HAGG);
    float* ecomb = (float*)(ws + WS_ECOMB);
    float* ncomb = (float*)(ws + WS_NCOMB);
    float* out = (float*)d_out;

    zero_kernel<<<4096, 256, 0, stream>>>(hagg, ecomb, ncomb);
    pack_kernel<<<1, 256, 0, stream>>>(We_ih, We_hh, be_ih, be_hh,
                                       Wn_ih, Wn_hh, bn_ih, bn_hh,
                                       Wu_ih, Wu_hh, bu_ih, bu_hh,
                                       fragE, fragN, fragU, biasE, biasN, biasU);
    edge_lstm_kernel<<<2048, 256, 0, stream>>>(edge_feat, node_feat, g_repr,
                                               edge_h1, edge_h2, edge_src, edge_dst,
                                               fragE, biasE, hagg, ecomb, out);
    node_lstm_kernel<<<128, 256, 0, stream>>>(node_feat, g_repr, node_h1, node_h2,
                                              hagg, fragN, biasN, ncomb, out);
    graph_lstm_kernel<<<1, 32, 0, stream>>>(g_repr, graph_h1, graph_h2,
                                            ecomb, ncomb, fragU, biasU, out);
}
